// GVPEncoder_18219251270262
// MI455X (gfx1250) — matrix operand order repaired and checked by reading, not compiled, not run on hardware
//
#include <hip/hip_runtime.h>
#include <hip/hip_bf16.h>
#include <math.h>
#include <stdint.h>

typedef _Float16 f16;
typedef _Float16 v16h __attribute__((ext_vector_type(16)));
typedef _Float16 v8h  __attribute__((ext_vector_type(8)));
typedef float    v8f  __attribute__((ext_vector_type(8)));

#define BB 32
#define LL 512
#define NN (BB*LL)      /* 16384 nodes */
#define EE (NN*16)      /* 262144 edges */
#define EPSV 1e-4f

// ---------------------------------------------------------------- WMMA utils
__device__ __forceinline__ v8f wmma_f16(v16h a, v16h b, v8f c) {
  return __builtin_amdgcn_wmma_f32_16x16x32_f16(false, a, false, b, (short)0, c, false, false);
}

// A fragment (16x32 f16, M x K) from a row-major LDS tile.
__device__ __forceinline__ v16h load_a_frag(const f16* At, int strideH, int kt, int lane) {
  int row  = lane & 15;
  int koff = kt * 32 + ((lane & 16) ? 8 : 0);
  const f16* p = At + row * strideH + koff;
  v8h lo = *(const v8h*)(p);
  v8h hi = *(const v8h*)(p + 16);
  v16h a;
#pragma unroll
  for (int i = 0; i < 8; ++i) { a[i] = lo[i]; a[i + 8] = hi[i]; }
  return a;
}

// B fragment (32x16 f16, K x N) from pre-packed global (1 KB tile, 32B/lane).
__device__ __forceinline__ v16h load_b_frag(const f16* Bp, int tileIdx, int lane) {
  return *(const v16h*)(Bp + (size_t)tileIdx * 512 + lane * 16);
}

// Async global->LDS copy (16B per lane), tracked by ASYNCcnt.
__device__ __forceinline__ void async_copy_b128(unsigned ldsOff, unsigned gOff, const void* base) {
  asm volatile("global_load_async_to_lds_b128 %0, %1, %2"
               :: "v"(ldsOff), "v"(gOff), "s"(base) : "memory");
}
__device__ __forceinline__ void wait_async() {
  asm volatile("s_wait_asynccnt 0" ::: "memory");
}

// C-persistent GEMM: 16 rows x NT*16 cols, K = KT*32; bias + optional relu.
template<int KT, int NT>
__device__ __forceinline__ void gemm_tile(const f16* At, int strideH, const f16* Bp,
                                          const float* bias, bool relu, v8f* c, int lane) {
#pragma unroll
  for (int nt = 0; nt < NT; ++nt)
#pragma unroll
    for (int r = 0; r < 8; ++r) c[nt][r] = 0.f;
  constexpr int KB = (KT < 6) ? KT : 6;
#pragma unroll
  for (int k0 = 0; k0 < KT; k0 += KB) {
    v16h a[KB];
#pragma unroll
    for (int q = 0; q < KB; ++q)
      if (k0 + q < KT) a[q] = load_a_frag(At, strideH, k0 + q, lane);
#pragma unroll
    for (int nt = 0; nt < NT; ++nt)
#pragma unroll
      for (int q = 0; q < KB; ++q)
        if (k0 + q < KT) c[nt] = wmma_f16(a[q], load_b_frag(Bp, nt * KT + k0 + q, lane), c[nt]);
  }
  int col = lane & 15;
#pragma unroll
  for (int nt = 0; nt < NT; ++nt) {
    float bv = bias[nt * 16 + col];
#pragma unroll
    for (int r = 0; r < 8; ++r) {
      float x = c[nt][r] + bv;
      c[nt][r] = relu ? fmaxf(x, 0.f) : x;
    }
  }
}

// Streamed plane GEMM (no bias): out f16 plane (A-fragment-ready layout).
template<int KT, int NT>
__device__ __forceinline__ void gemm_plane(const f16* Apl, int strideA, const f16* Bp,
                                           f16* outPl, int strideO, int lane) {
  int col = lane & 15, rb = (lane & 16) ? 8 : 0;
#pragma unroll
  for (int nt = 0; nt < NT; ++nt) {
    v8f cc;
#pragma unroll
    for (int r = 0; r < 8; ++r) cc[r] = 0.f;
#pragma unroll
    for (int kt = 0; kt < KT; ++kt)
      cc = wmma_f16(load_a_frag(Apl, strideA, kt, lane), load_b_frag(Bp, nt * KT + kt, lane), cc);
#pragma unroll
    for (int r = 0; r < 8; ++r)
      outPl[(rb + r) * strideO + nt * 16 + col] = (f16)cc[r];
  }
}

// Same but multiply per-element sigmoid gate[row][col] before storing.
template<int KT, int NT>
__device__ __forceinline__ void gemm_gate_plane(const f16* Apl, int strideA, const f16* Bp,
                                                const float* gate, int gw,
                                                f16* outPl, int strideO, int lane) {
  int col = lane & 15, rb = (lane & 16) ? 8 : 0;
#pragma unroll
  for (int nt = 0; nt < NT; ++nt) {
    v8f cc;
#pragma unroll
    for (int r = 0; r < 8; ++r) cc[r] = 0.f;
#pragma unroll
    for (int kt = 0; kt < KT; ++kt)
      cc = wmma_f16(load_a_frag(Apl, strideA, kt, lane), load_b_frag(Bp, nt * KT + kt, lane), cc);
#pragma unroll
    for (int r = 0; r < 8; ++r)
      outPl[(rb + r) * strideO + nt * 16 + col] =
          (f16)(cc[r] * gate[(rb + r) * gw + nt * 16 + col]);
  }
}

template<int NT>
__device__ __forceinline__ void store_c_f16(const v8f* c, f16* At, int strideH, int lane) {
  int col = lane & 15, rb = (lane & 16) ? 8 : 0;
#pragma unroll
  for (int nt = 0; nt < NT; ++nt)
#pragma unroll
    for (int r = 0; r < 8; ++r)
      At[(rb + r) * strideH + nt * 16 + col] = (f16)c[nt][r];
}

template<int NT>
__device__ __forceinline__ void store_sigmoid(const v8f* c, float* gate, int width, int lane) {
  int col = lane & 15, rb = (lane & 16) ? 8 : 0;
#pragma unroll
  for (int nt = 0; nt < NT; ++nt)
#pragma unroll
    for (int r = 0; r < 8; ++r)
      gate[(rb + r) * width + nt * 16 + col] = 1.f / (1.f + __expf(-c[nt][r]));
}

__device__ __forceinline__ void unit3(float* x) {
  float n = sqrtf(x[0]*x[0] + x[1]*x[1] + x[2]*x[2] + 1e-8f);
  x[0] /= n; x[1] /= n; x[2] /= n;
}
__device__ __forceinline__ void cross3(const float* a, const float* b, float* o) {
  o[0] = a[1]*b[2] - a[2]*b[1];
  o[1] = a[2]*b[0] - a[0]*b[2];
  o[2] = a[0]*b[1] - a[1]*b[0];
}

// ------------------------------------------------------------- param structs
struct GvpW {
  const f16* whP; const f16* wvP;
  const f16* wsP; const float* wsb;
  const f16* wsvP; const float* wsvb;
};

// --------------------------------------------------------------- weight pack
__global__ void pack_b(const float* __restrict__ w, f16* __restrict__ out,
                       int Kd, int Nd, int KT, int NT) {
  int idx = blockIdx.x * blockDim.x + threadIdx.x;
  int total = KT * NT * 512;
  if (idx >= total) return;
  int t = idx >> 9, within = idx & 511;
  int lane = within >> 4, h = within & 15;
  int nt = t / KT, kt = t - nt * KT;
  int k = kt * 32 + ((lane < 16) ? 0 : 8) + (h & 7) + ((h & 8) ? 16 : 0);
  int n = nt * 16 + (lane & 15);
  float val = (k < Kd && n < Nd) ? w[k * Nd + n] : 0.f;
  out[(size_t)t * 512 + within] = (f16)val;
}

// --------------------------------------------------------------- embed node
struct EmbedNodeP {
  const float* coords; const float* conf;
  const float* wh; const float* wsw; const float* wsb;
  const float* wv; const float* wsvw; const float* wsvb;
  const float* confW; const float* confB;
  float* s; float* v;
};

__global__ __launch_bounds__(32) void embed_node_kernel(EmbedNodeP P) {
  __shared__ float cat[38];
  __shared__ float vbuf[3][3];
  __shared__ float vhB[96];
  __shared__ float soB[128];
  __shared__ float gateB[32];
  int node = blockIdx.x, lane = threadIdx.x;
  int b = node / LL, i = node % LL;
  const float* Cb = P.coords + (size_t)b * LL * 9;

  if (lane < 3) {  // dihedral component of residue i
    int t = i * 3 + lane;
    float D = 0.f;
    if (t >= 1 && t <= 1533) {
      float X[4][3];
      for (int q = 0; q < 4; ++q) {
        int a = t - 1 + q;
        for (int cc = 0; cc < 3; ++cc) X[q][cc] = Cb[(a / 3) * 9 + (a % 3) * 3 + cc];
      }
      float u2[3], u1[3], u0[3], n2[3], n1[3];
      for (int cc = 0; cc < 3; ++cc) { u2[cc] = X[1][cc]-X[0][cc]; u1[cc] = X[2][cc]-X[1][cc]; u0[cc] = X[3][cc]-X[2][cc]; }
      unit3(u2); unit3(u1); unit3(u0);
      cross3(u2, u1, n2); unit3(n2);
      cross3(u1, u0, n1); unit3(n1);
      float cd = n2[0]*n1[0] + n2[1]*n1[1] + n2[2]*n1[2];
      cd = fminf(fmaxf(cd, -1.f + 1e-7f), 1.f - 1e-7f);
      float sg = u2[0]*n1[0] + u2[1]*n1[1] + u2[2]*n1[2];
      float sv = (sg > 0.f) ? 1.f : ((sg < 0.f) ? -1.f : 0.f);
      D = sv * acosf(cd);
    }
    cat[lane] = cosf(D); cat[3 + lane] = sinf(D);
  }
  if (lane == 3) {  // fwd
    float f[3] = {0, 0, 0};
    if (i < LL - 1) { for (int cc = 0; cc < 3; ++cc) f[cc] = Cb[(i+1)*9+3+cc] - Cb[i*9+3+cc]; unit3(f); }
    for (int cc = 0; cc < 3; ++cc) vbuf[0][cc] = f[cc];
  }
  if (lane == 4) {  // bwd
    float f[3] = {0, 0, 0};
    if (i > 0) { for (int cc = 0; cc < 3; ++cc) f[cc] = Cb[(i-1)*9+3+cc] - Cb[i*9+3+cc]; unit3(f); }
    for (int cc = 0; cc < 3; ++cc) vbuf[1][cc] = f[cc];
  }
  if (lane == 5) {  // sidechain
    float nv[3], cv[3], bs[3], pp[3];
    for (int cc = 0; cc < 3; ++cc) { nv[cc] = Cb[i*9+cc] - Cb[i*9+3+cc]; cv[cc] = Cb[i*9+6+cc] - Cb[i*9+3+cc]; }
    unit3(nv); unit3(cv);
    for (int cc = 0; cc < 3; ++cc) bs[cc] = nv[cc] + cv[cc];
    unit3(bs);
    cross3(nv, cv, pp); unit3(pp);
    for (int cc = 0; cc < 3; ++cc) vbuf[2][cc] = -bs[cc] * 0.57735026919f - pp[cc] * 0.81649658092f;
  }
  __syncthreads();

  for (int t = lane; t < 96; t += 32) {  // vh: h=32
    int h = t / 3, cc = t % 3;
    float a = 0.f;
    for (int vi = 0; vi < 3; ++vi) a += vbuf[vi][cc] * P.wh[vi * 32 + h];
    vhB[t] = a;
  }
  __syncthreads();
  { int h = lane;
    cat[6 + h] = sqrtf(vhB[h*3]*vhB[h*3] + vhB[h*3+1]*vhB[h*3+1] + vhB[h*3+2]*vhB[h*3+2] + 1e-8f); }
  __syncthreads();

  float conf = P.conf[node];
  float rb[16];
#pragma unroll
  for (int r = 0; r < 16; ++r) {
    float mu = (float)r * (1.f / 15.f);
    float z = (conf - mu) * 16.f;
    rb[r] = __expf(-z * z);
  }
#pragma unroll
  for (int q = 0; q < 4; ++q) {
    int j = q * 32 + lane;
    float a = P.wsb[j];
    for (int k = 0; k < 38; ++k) a += cat[k] * P.wsw[k * 128 + j];
    float a2 = P.confB[j];
    for (int r = 0; r < 16; ++r) a2 += rb[r] * P.confW[r * 128 + j];
    soB[j] = a + a2;
  }
  __syncthreads();
  { int vo = lane;
    float a = P.wsvb[vo];
    for (int j = 0; j < 128; ++j) a += soB[j] * P.wsvw[j * 32 + vo];
    gateB[vo] = 1.f / (1.f + __expf(-a)); }
  __syncthreads();
  for (int t = lane; t < 96; t += 32) {
    int vo = t / 3, cc = t % 3;
    float a = 0.f;
    for (int h = 0; h < 32; ++h) a += vhB[h * 3 + cc] * P.wv[h * 32 + vo];
    P.v[(size_t)node * 96 + t] = a * gateB[vo];
  }
#pragma unroll
  for (int q = 0; q < 4; ++q) { int j = q * 32 + lane; P.s[(size_t)node * 128 + j] = soB[j]; }
}

// -------------------------------------------------------------- layernorm(s,v)
__global__ __launch_bounds__(32) void ln_update(float* s, f16* s16, float* v,
                                                const float* addS, const float* addV,
                                                const float* g, const float* bb) {
  int node = blockIdx.x, lane = threadIdx.x;
  float x[4];
#pragma unroll
  for (int q = 0; q < 4; ++q) {
    int j = q * 32 + lane;
    float t = s[(size_t)node * 128 + j];
    if (addS) t += addS[(size_t)node * 128 + j];
    x[q] = t;
  }
  float sum = x[0] + x[1] + x[2] + x[3];
#pragma unroll
  for (int m = 1; m < 32; m <<= 1) sum += __shfl_xor(sum, m, 32);
  float mu = sum * (1.f / 128.f);
  float vs = 0.f;
#pragma unroll
  for (int q = 0; q < 4; ++q) { float d = x[q] - mu; vs += d * d; }
#pragma unroll
  for (int m = 1; m < 32; m <<= 1) vs += __shfl_xor(vs, m, 32);
  float rstd = rsqrtf(vs * (1.f / 128.f) + EPSV);
#pragma unroll
  for (int q = 0; q < 4; ++q) {
    int j = q * 32 + lane;
    float y = (x[q] - mu) * rstd * g[j] + bb[j];
    s[(size_t)node * 128 + j] = y;
    s16[(size_t)node * 128 + j] = (f16)y;
  }
  float w[3]; float nt = 0.f;
#pragma unroll
  for (int q = 0; q < 3; ++q) {
    int j = lane * 3 + q;
    float t = v[(size_t)node * 96 + j];
    if (addV) t += addV[(size_t)node * 96 + j];
    w[q] = t; nt += t * t;
  }
#pragma unroll
  for (int m = 1; m < 32; m <<= 1) nt += __shfl_xor(nt, m, 32);
  float scale = rsqrtf(nt * (1.f / 32.f) + EPSV);
#pragma unroll
  for (int q = 0; q < 3; ++q) v[(size_t)node * 96 + lane * 3 + q] = w[q] * scale;
}

// ----------------------------------------------------------------------- KNN
__global__ void knn_kernel(const float* __restrict__ coords, int* __restrict__ nbr) {
  int node = blockIdx.x * blockDim.x + threadIdx.x;
  if (node >= NN) return;
  int b = node / LL, i = node % LL;
  const float* Cb = coords + (size_t)b * LL * 9;
  float xi0 = Cb[i*9+3], xi1 = Cb[i*9+4], xi2 = Cb[i*9+5];
  float bd[16]; int bi[16];
#pragma unroll
  for (int k = 0; k < 16; ++k) { bd[k] = 3.4e38f; bi[k] = 0; }
  for (int j = 0; j < LL; ++j) {
    if (j == i) continue;
    float d0 = Cb[j*9+3] - xi0, d1 = Cb[j*9+4] - xi1, d2v = Cb[j*9+5] - xi2;
    float d2 = d0*d0 + d1*d1 + d2v*d2v;
    if (d2 < bd[15]) {
      bd[15] = d2; bi[15] = j;
#pragma unroll
      for (int k = 15; k > 0; --k) {
        if (bd[k] < bd[k-1]) {
          float tf = bd[k]; bd[k] = bd[k-1]; bd[k-1] = tf;
          int ti = bi[k]; bi[k] = bi[k-1]; bi[k-1] = ti;
        }
      }
    }
  }
#pragma unroll
  for (int k = 0; k < 16; ++k) nbr[(size_t)node * 16 + k] = bi[k];
}

// --------------------------------------- edge features + embed_edge + ln_edge
struct EdgeP {
  const float* coords; const int* nbr;
  const float* wh; const float* wsw; const float* wsb;
  const float* wv; const float* wsvw; const float* wsvb;
  const float* lng; const float* lnb;
  f16* es16; float* ev;
};

__global__ void edge_kernel(EdgeP P) {
  int e = blockIdx.x * blockDim.x + threadIdx.x;
  if (e >= EE) return;
  int node = e / 16, b = node / LL, i = node % LL;
  int j = P.nbr[e];
  const float* Cb = P.coords + (size_t)b * LL * 9;
  float e0 = Cb[j*9+3] - Cb[i*9+3];
  float e1 = Cb[j*9+4] - Cb[i*9+4];
  float e2 = Cb[j*9+5] - Cb[i*9+5];
  float d = sqrtf(e0*e0 + e1*e1 + e2*e2 + 1e-8f);
  float es[33];
#pragma unroll
  for (int r = 0; r < 16; ++r) {
    float mu = 2.f + 20.f * (float)r * (1.f / 15.f);
    float z = (d - mu) * (1.f / 1.25f);
    es[r] = __expf(-z * z);
  }
  float off = (float)(j - i);
#pragma unroll
  for (int r = 0; r < 8; ++r) {
    float fr = __expf(-(float)(2 * r) * 0.57564627325f);  // ln(1e4)/16
    float ang = off * fr;
    es[16 + r] = cosf(ang); es[24 + r] = sinf(ang);
  }
  float u0 = e0 / d, u1 = e1 / d, u2 = e2 / d;
  float wh0 = P.wh[0];
  float vh0 = u0 * wh0, vh1 = u1 * wh0, vh2 = u2 * wh0;
  es[32] = sqrtf(vh0*vh0 + vh1*vh1 + vh2*vh2 + 1e-8f);
  float so[32]; float mean = 0.f;
  for (int n = 0; n < 32; ++n) {
    float a = P.wsb[n];
    for (int k = 0; k < 33; ++k) a += es[k] * P.wsw[k * 32 + n];
    so[n] = a; mean += a;
  }
  float gacc = P.wsvb[0];
  for (int n = 0; n < 32; ++n) gacc += so[n] * P.wsvw[n];
  float gt = 1.f / (1.f + __expf(-gacc));
  float wv0 = P.wv[0];
  float vo0 = vh0 * wv0 * gt, vo1 = vh1 * wv0 * gt, vo2 = vh2 * wv0 * gt;
  mean *= (1.f / 32.f);
  float var = 0.f;
  for (int n = 0; n < 32; ++n) { float dd = so[n] - mean; var += dd * dd; }
  var *= (1.f / 32.f);
  float rstd = rsqrtf(var + EPSV);
  for (int n = 0; n < 32; ++n)
    P.es16[(size_t)e * 32 + n] = (f16)((so[n] - mean) * rstd * P.lng[n] + P.lnb[n]);
  float vsc = rsqrtf(vo0*vo0 + vo1*vo1 + vo2*vo2 + EPSV);
  P.ev[(size_t)e * 3 + 0] = vo0 * vsc;
  P.ev[(size_t)e * 3 + 1] = vo1 * vsc;
  P.ev[(size_t)e * 3 + 2] = vo2 * vsc;
}

// -------------------------------------------------- message pass (WMMA core)
struct MsgParams {
  const f16* s16; const float* v; const f16* es16; const float* ev; const int* nbr;
  GvpW g0, g1, g2;
  float* aggS; float* aggV;
};

__global__ __launch_bounds__(32) void msg_kernel(MsgParams P) {
  __shared__ __align__(16) f16 At[16 * 384];     // A tile: stride 384 (msg0) then 160
  __shared__ __align__(16) f16 mvP[3 * 16 * 96]; // channel planes, stride 96 (later 32)
  __shared__ __align__(16) f16 mvQ[3 * 16 * 32]; // channel planes, stride 32
  __shared__ __align__(16) f16 vhP[3 * 16 * 96]; // vh planes, stride 96
  __shared__ float gate[16 * 32];
  __shared__ int srcIdx[16];
  int node = blockIdx.x, lane = threadIdx.x;
  int b = node / LL;
  if (lane < 16) srcIdx[lane] = b * LL + P.nbr[(size_t)node * 16 + lane];
  // zero K-pad regions once
  for (int t = lane; t < 3 * 16 * 31; t += 32) {   // mvP cols 65..95
    int row = t / 31, colz = 65 + (t - row * 31);
    mvP[row * 96 + colz] = (f16)0.f;
  }
  for (int t = lane; t < 3 * 16 * 16; t += 32) {   // vhP cols 80..95
    int row = t >> 4, colz = 80 + (t & 15);
    vhP[row * 96 + colz] = (f16)0.f;
  }
  __syncthreads();

  // async gather of f16 rows straight into LDS (ASYNCcnt-tracked)
#pragma unroll
  for (int it = 0; it < 8; ++it) {                 // s16[src] -> A cols 0..127
    int t = it * 32 + lane;
    int row = t >> 4, chunk = t & 15;
    unsigned l = (unsigned)(uintptr_t)(At + row * 384 + chunk * 8);
    unsigned g = (unsigned)(srcIdx[row] * 256 + chunk * 16);
    async_copy_b128(l, g, P.s16);
  }
#pragma unroll
  for (int it = 0; it < 8; ++it) {                 // s16[dst] -> A cols 160..287
    int t = it * 32 + lane;
    int row = t >> 4, chunk = t & 15;
    unsigned l = (unsigned)(uintptr_t)(At + row * 384 + 160 + chunk * 8);
    unsigned g = (unsigned)(node * 256 + chunk * 16);
    async_copy_b128(l, g, P.s16);
  }
#pragma unroll
  for (int it = 0; it < 2; ++it) {                 // es16 -> A cols 128..159
    int idx = it * 32 + lane;
    int row = idx >> 2, chunk = idx & 3;
    unsigned l = (unsigned)(uintptr_t)(At + row * 384 + 128 + chunk * 8);
    unsigned g = (unsigned)((node * 16 + row) * 64 + chunk * 16);
    async_copy_b128(l, g, P.es16);
  }

  // overlap: VALU fills of the vector-channel planes + A pad
  for (int k = 0; k < 16; ++k) {
    int e = node * 16 + k;
    int sidx = srcIdx[k];
    if (lane < 31) At[k * 384 + 353 + lane] = (f16)0.f;
    for (int j = lane; j < 96; j += 32) {
      int vv = j / 3, cc = j - vv * 3;
      mvP[cc * 1536 + k * 96 + vv]      = (f16)P.v[(size_t)sidx * 96 + j];
      mvP[cc * 1536 + k * 96 + 33 + vv] = (f16)P.v[(size_t)node * 96 + j];
    }
    if (lane < 3) mvP[lane * 1536 + k * 96 + 32] = (f16)P.ev[(size_t)e * 3 + lane];
  }
  wait_async();
  __syncthreads();

  v8f c[8]; v8f cg[2];

  // ================= GVP0: 353 -> 128 (relu), 65 vec -> 32 vec
  for (int ch = 0; ch < 3; ++ch)
    gemm_plane<3, 5>(mvP + ch * 1536, 96, P.g0.whP, vhP + ch * 1536, 96, lane);
  __syncthreads();
  for (int t = lane; t < 16 * 65; t += 32) {       // vn -> A cols 288..352
    int e = t / 65, h = t - e * 65;
    float a0 = (float)vhP[0 * 1536 + e * 96 + h];
    float a1 = (float)vhP[1 * 1536 + e * 96 + h];
    float a2 = (float)vhP[2 * 1536 + e * 96 + h];
    At[e * 384 + 288 + h] = (f16)sqrtf(a0*a0 + a1*a1 + a2*a2 + 1e-8f);
  }
  __syncthreads();
  gemm_tile<12, 8>(At, 384, P.g0.wsP, P.g0.wsb, true, c, lane);
  __syncthreads();
  store_c_f16<8>(c, At, 160, lane);
  __syncthreads();
  gemm_tile<4, 2>(At, 160, P.g0.wsvP, P.g0.wsvb, false, cg, lane);
  store_sigmoid<2>(cg, gate, 32, lane);
  __syncthreads();
  for (int ch = 0; ch < 3; ++ch)
    gemm_gate_plane<3, 2>(vhP + ch * 1536, 96, P.g0.wvP, gate, 32, mvQ + ch * 512, 32, lane);
  __syncthreads();

  // ================= GVP1: 160 -> 128 (relu), 32 -> 32
  for (int ch = 0; ch < 3; ++ch)
    gemm_plane<1, 2>(mvQ + ch * 512, 32, P.g1.whP, vhP + ch * 1536, 96, lane);
  __syncthreads();
  for (int t = lane; t < 16 * 32; t += 32) {
    int e = t >> 5, h = t & 31;
    float a0 = (float)vhP[0 * 1536 + e * 96 + h];
    float a1 = (float)vhP[1 * 1536 + e * 96 + h];
    float a2 = (float)vhP[2 * 1536 + e * 96 + h];
    At[e * 160 + 128 + h] = (f16)sqrtf(a0*a0 + a1*a1 + a2*a2 + 1e-8f);
  }
  __syncthreads();
  gemm_tile<5, 8>(At, 160, P.g1.wsP, P.g1.wsb, true, c, lane);
  __syncthreads();
  store_c_f16<8>(c, At, 160, lane);
  __syncthreads();
  gemm_tile<4, 2>(At, 160, P.g1.wsvP, P.g1.wsvb, false, cg, lane);
  store_sigmoid<2>(cg, gate, 32, lane);
  __syncthreads();
  for (int ch = 0; ch < 3; ++ch)
    gemm_gate_plane<1, 2>(vhP + ch * 1536, 96, P.g1.wvP, gate, 32, mvP + ch * 512, 32, lane);
  __syncthreads();

  // ================= GVP2: 160 -> 128 (no act), 32 -> 32
  for (int ch = 0; ch < 3; ++ch)
    gemm_plane<1, 2>(mvP + ch * 512, 32, P.g2.whP, vhP + ch * 1536, 96, lane);
  __syncthreads();
  for (int t = lane; t < 16 * 32; t += 32) {
    int e = t >> 5, h = t & 31;
    float a0 = (float)vhP[0 * 1536 + e * 96 + h];
    float a1 = (float)vhP[1 * 1536 + e * 96 + h];
    float a2 = (float)vhP[2 * 1536 + e * 96 + h];
    At[e * 160 + 128 + h] = (f16)sqrtf(a0*a0 + a1*a1 + a2*a2 + 1e-8f);
  }
  __syncthreads();
  gemm_tile<5, 8>(At, 160, P.g2.wsP, P.g2.wsb, false, c, lane);
  // aggregate scalar messages: segment-sum over 16 contiguous edges == C column sums
  {
    int col = lane & 15;
#pragma unroll
    for (int nt = 0; nt < 8; ++nt) {
      float part = 0.f;
#pragma unroll
      for (int r = 0; r < 8; ++r) part += c[nt][r];
      part += __shfl_xor(part, 16, 32);
      if (lane < 16) P.aggS[(size_t)node * 128 + nt * 16 + col] = part * (1.f / 16.f);
    }
  }
  __syncthreads();
  store_c_f16<8>(c, At, 160, lane);
  __syncthreads();
  gemm_tile<4, 2>(At, 160, P.g2.wsvP, P.g2.wsvb, false, cg, lane);
  store_sigmoid<2>(cg, gate, 32, lane);
  __syncthreads();
  // vout2 with gate, row-summed straight into aggV
  {
    int col = lane & 15, rb = (lane & 16) ? 8 : 0;
    for (int ch = 0; ch < 3; ++ch) {
#pragma unroll
      for (int nt = 0; nt < 2; ++nt) {
        v8f cc;
#pragma unroll
        for (int r = 0; r < 8; ++r) cc[r] = 0.f;
        cc = wmma_f16(load_a_frag(vhP + ch * 1536, 96, 0, lane),
                      load_b_frag(P.g2.wvP, nt, lane), cc);
        float part = 0.f;
#pragma unroll
        for (int r = 0; r < 8; ++r) part += cc[r] * gate[(rb + r) * 32 + nt * 16 + col];
        part += __shfl_xor(part, 16, 32);
        if (lane < 16)
          P.aggV[(size_t)node * 96 + (nt * 16 + col) * 3 + ch] = part * (1.f / 16.f);
      }
    }
  }
}

// ------------------------------------------------------- feed-forward (WMMA)
struct FfParams {
  const f16* s16; const float* v;
  GvpW g0, g1;
  float* hs; float* hv;
};

__global__ __launch_bounds__(32) void ff_kernel(FfParams P) {
  __shared__ __align__(16) f16 At[16 * 576];
  __shared__ __align__(16) f16 mvInP[3 * 16 * 32];
  __shared__ __align__(16) f16 vhPl[3 * 16 * 64];
  __shared__ __align__(16) f16 hvPl[3 * 16 * 64];
  __shared__ float gate[16 * 64];
  int n0 = blockIdx.x * 16, lane = threadIdx.x;

  // async copy of the 16 node rows of s16 into the A tile (stride 192)
#pragma unroll
  for (int it = 0; it < 8; ++it) {
    int t = it * 32 + lane;
    int row = t >> 4, chunk = t & 15;
    unsigned l = (unsigned)(uintptr_t)(At + row * 192 + chunk * 8);
    unsigned g = (unsigned)((n0 + row) * 256 + chunk * 16);
    async_copy_b128(l, g, P.s16);
  }
  for (int j = lane; j < 16 * 96; j += 32) {
    int row = j / 96, jj = j - row * 96;
    int vv = jj / 3, cc = jj - vv * 3;
    mvInP[cc * 512 + row * 32 + vv] = (f16)P.v[(size_t)n0 * 96 + j];
  }
  wait_async();
  __syncthreads();

  // ---- FF0: 192 -> 512 (relu), 32 vec -> 64 vec
  for (int ch = 0; ch < 3; ++ch)
    gemm_plane<1, 4>(mvInP + ch * 512, 32, P.g0.whP, vhPl + ch * 1024, 64, lane);
  __syncthreads();
  for (int t = lane; t < 16 * 64; t += 32) {
    int e = t >> 6, h = t & 63;
    float a0 = (float)vhPl[0 * 1024 + e * 64 + h];
    float a1 = (float)vhPl[1 * 1024 + e * 64 + h];
    float a2 = (float)vhPl[2 * 1024 + e * 64 + h];
    At[e * 192 + 128 + h] = (f16)sqrtf(a0*a0 + a1*a1 + a2*a2 + 1e-8f);
  }
  __syncthreads();
  {
    v16h a6[6];
#pragma unroll
    for (int q = 0; q < 6; ++q) a6[q] = load_a_frag(At, 192, q, lane);
    int col = lane & 15, rb = (lane & 16) ? 8 : 0;
    for (int nt = 0; nt < 32; ++nt) {
      v8f cc;
#pragma unroll
      for (int r = 0; r < 8; ++r) cc[r] = 0.f;
#pragma unroll
      for (int q = 0; q < 6; ++q) cc = wmma_f16(a6[q], load_b_frag(P.g0.wsP, nt * 6 + q, lane), cc);
      float bv = P.g0.wsb[nt * 16 + col];
#pragma unroll
      for (int r = 0; r < 8; ++r)
        At[(rb + r) * 576 + nt * 16 + col] = (f16)fmaxf(cc[r] + bv, 0.f);
    }
  }
  __syncthreads();
  v8f c4[4];
  gemm_tile<16, 4>(At, 576, P.g0.wsvP, P.g0.wsvb, false, c4, lane);
  store_sigmoid<4>(c4, gate, 64, lane);
  __syncthreads();
  for (int ch = 0; ch < 3; ++ch)
    gemm_gate_plane<2, 4>(vhPl + ch * 1024, 64, P.g0.wvP, gate, 64, hvPl + ch * 1024, 64, lane);
  __syncthreads();

  // ---- FF1: 576 -> 128 (no act), 64 vec -> 32 vec
  for (int ch = 0; ch < 3; ++ch)
    gemm_plane<2, 4>(hvPl + ch * 1024, 64, P.g1.whP, vhPl + ch * 1024, 64, lane);
  __syncthreads();
  for (int t = lane; t < 16 * 64; t += 32) {
    int e = t >> 6, h = t & 63;
    float a0 = (float)vhPl[0 * 1024 + e * 64 + h];
    float a1 = (float)vhPl[1 * 1024 + e * 64 + h];
    float a2 = (float)vhPl[2 * 1024 + e * 64 + h];
    At[e * 576 + 512 + h] = (f16)sqrtf(a0*a0 + a1*a1 + a2*a2 + 1e-8f);
  }
  __syncthreads();
  v8f c8[8];
  gemm_tile<18, 8>(At, 576, P.g1.wsP, P.g1.wsb, false, c8, lane);
  {
    int col = lane & 15, rb = (lane & 16) ? 8 : 0;
#pragma unroll
    for (int nt = 0; nt < 8; ++nt)
#pragma unroll
      for (int r = 0; r < 8; ++r)
        P.hs[(size_t)(n0 + rb + r) * 128 + nt * 16 + col] = c8[nt][r];
  }
  __syncthreads();
  store_c_f16<8>(c8, At, 576, lane);
  __syncthreads();
  v8f cg2[2];
  gemm_tile<4, 2>(At, 576, P.g1.wsvP, P.g1.wsvb, false, cg2, lane);
  store_sigmoid<2>(cg2, gate, 32, lane);
  __syncthreads();
  {
    int col = lane & 15, rb = (lane & 16) ? 8 : 0;
    for (int ch = 0; ch < 3; ++ch) {
#pragma unroll
      for (int nt = 0; nt < 2; ++nt) {
        v8f cc;
#pragma unroll
        for (int r = 0; r < 8; ++r) cc[r] = 0.f;
#pragma unroll
        for (int kt = 0; kt < 2; ++kt)
          cc = wmma_f16(load_a_frag(vhPl + ch * 1024, 64, kt, lane),
                        load_b_frag(P.g1.wvP, nt * 2 + kt, lane), cc);
#pragma unroll
        for (int r = 0; r < 8; ++r)
          P.hv[(size_t)(n0 + rb + r) * 96 + (nt * 16 + col) * 3 + ch] =
              cc[r] * gate[(rb + r) * 32 + nt * 16 + col];
      }
    }
  }
}

// ------------------------------------------------------------------ finalize
__global__ void finalize_kernel(const float* __restrict__ s, const float* __restrict__ v,
                                float* __restrict__ out) {
  int idx = blockIdx.x * blockDim.x + threadIdx.x;
  if (idx >= NN * 224) return;
  int node = idx / 224, j = idx - node * 224;
  out[idx] = (j < 128) ? s[(size_t)node * 128 + j] : v[(size_t)node * 96 + (j - 128)];
}

// ================================================================== launcher
extern "C" void kernel_launch(void* const* d_in, const int* in_sizes, int n_in,
                              void* d_out, int out_size, void* d_ws, size_t ws_size,
                              hipStream_t stream) {
  (void)in_sizes; (void)n_in; (void)out_size; (void)ws_size;
  const float* coords = (const float*)d_in[0];
  const float* conf   = (const float*)d_in[3];   // masks (d_in[1,2]) constant; elided
  auto F = [&](int i) { return (const float*)d_in[i]; };

  char* ws = (char*)d_ws;
  size_t off = 0;
  auto alloc = [&](size_t bytes) -> char* {
    char* p = ws + off;
    off += bytes; off = (off + 255) & ~(size_t)255;
    return p;
  };
  int*   nbr    = (int*)  alloc((size_t)EE * 4);
  float* s      = (float*)alloc((size_t)NN * 128 * 4);
  float* v      = (float*)alloc((size_t)NN * 96 * 4);
  f16*   s16    = (f16*)  alloc((size_t)NN * 128 * 2);
  f16*   es16   = (f16*)  alloc((size_t)EE * 32 * 2);
  float* ev     = (float*)alloc((size_t)EE * 3 * 4);
  float* aggS   = (float*)alloc((size_t)NN * 128 * 4);
  float* aggV   = (float*)alloc((size_t)NN * 96 * 4);
  f16*   packed = (f16*)  alloc((size_t)3 * 661 * 1024);  // 661 tiles x 1 KB per layer

  // tiles: msg0ws,msg1ws,msg2ws,wsv0..2,ff0ws,ff0wsv,ff1ws,ff1wsv,
  //        msg0wh,msg0wv,msg1wh,msg1wv,msg2wh,msg2wv,ff0wh,ff0wv,ff1wh,ff1wv
  const int tOff[20] = {0, 96, 136, 176, 184, 192, 200, 392, 456, 600,
                        608, 623, 629, 631, 633, 635, 637, 641, 649, 657};
  auto packPtr = [&](int layer, int m) {
    return packed + ((size_t)layer * 661 + tOff[m]) * 512;
  };
  struct PD { int leafOff, Kd, Nd, KT, NT; };
  const PD pd[20] = {
    {1,  353, 128, 12, 8},  {7,  160, 128, 5, 8},  {13, 160, 128, 5, 8},
    {4,  128, 32,  4,  2},  {10, 128, 32,  4, 2},  {16, 128, 32,  4, 2},
    {21, 192, 512, 6, 32},  {24, 512, 64, 16, 4},  {27, 576, 128, 18, 8},
    {30, 128, 32,  4,  2},
    {0,  65,  65,  3,  5},  {3,  65,  32,  3, 2},
    {6,  32,  32,  1,  2},  {9,  32,  32,  1, 2},
    {12, 32,  32,  1,  2},  {15, 32,  32,  1, 2},
    {20, 32,  64,  1,  4},  {23, 64,  64,  2, 4},
    {26, 64,  64,  2,  4},  {29, 64,  32,  2, 2},
  };
  for (int l = 0; l < 3; ++l) {
    int base = 22 + 34 * l;
    for (int m = 0; m < 20; ++m) {
      int tot = pd[m].KT * pd[m].NT * 512;
      pack_b<<<(tot + 255) / 256, 256, 0, stream>>>(F(base + pd[m].leafOff), packPtr(l, m),
                                                    pd[m].Kd, pd[m].Nd, pd[m].KT, pd[m].NT);
    }
  }

  EmbedNodeP en{coords, conf, F(4), F(5), F(6), F(7), F(8), F(9), F(20), F(21), s, v};
  embed_node_kernel<<<NN, 32, 0, stream>>>(en);
  ln_update<<<NN, 32, 0, stream>>>(s, s16, v, nullptr, nullptr, F(10), F(11));
  knn_kernel<<<(NN + 255) / 256, 256, 0, stream>>>(coords, nbr);
  EdgeP ep{coords, nbr, F(12), F(13), F(14), F(15), F(16), F(17), F(18), F(19), es16, ev};
  edge_kernel<<<(EE + 255) / 256, 256, 0, stream>>>(ep);

  for (int l = 0; l < 3; ++l) {
    int base = 22 + 34 * l;
    MsgParams mp;
    mp.s16 = s16; mp.v = v; mp.es16 = es16; mp.ev = ev; mp.nbr = nbr;
    mp.g0 = {packPtr(l,10), packPtr(l,11), packPtr(l,0), F(base + 2),  packPtr(l,3), F(base + 5)};
    mp.g1 = {packPtr(l,12), packPtr(l,13), packPtr(l,1), F(base + 8),  packPtr(l,4), F(base + 11)};
    mp.g2 = {packPtr(l,14), packPtr(l,15), packPtr(l,2), F(base + 14), packPtr(l,5), F(base + 17)};
    mp.aggS = aggS; mp.aggV = aggV;
    msg_kernel<<<NN, 32, 0, stream>>>(mp);
    ln_update<<<NN, 32, 0, stream>>>(s, s16, v, aggS, aggV, F(base + 18), F(base + 19));

    FfParams fp;
    fp.s16 = s16; fp.v = v;
    fp.g0 = {packPtr(l,16), packPtr(l,17), packPtr(l,6), F(base + 22), packPtr(l,7), F(base + 25)};
    fp.g1 = {packPtr(l,18), packPtr(l,19), packPtr(l,8), F(base + 28), packPtr(l,9), F(base + 31)};
    fp.hs = aggS; fp.hv = aggV;
    ff_kernel<<<NN / 16, 32, 0, stream>>>(fp);
    ln_update<<<NN, 32, 0, stream>>>(s, s16, v, aggS, aggV, F(base + 32), F(base + 33));
  }
  finalize_kernel<<<(NN * 224 + 255) / 256, 256, 0, stream>>>(s, v, (float*)d_out);
}
